// Molecule3DBias_35631048688001
// MI455X (gfx1250) — matrix operand order repaired and checked by reading, not compiled, not run on hardware
//
#include <hip/hip_runtime.h>
#include <hip/hip_bf16.h>
#include <math.h>

typedef __attribute__((ext_vector_type(16))) _Float16 v16h;
typedef __attribute__((ext_vector_type(8)))  _Float16 v8h;
typedef __attribute__((ext_vector_type(8)))  float    v8f;

#define BB   8
#define NN   256
#define KK   128
#define HH   32
#define EMB  768
#define ALPHA 1.0f      /* 5/(5-0) */
#define EFS  136        /* padded LDS row stride in halves (kills bank conflicts) */

// Branchless exact-GELU: erf via Abramowitz-Stegun 7.1.26 (|err| < 1.5e-7),
// using hardware v_rcp_f32 / v_exp_f32. No divergent paths.
__device__ __forceinline__ float gelu_exact(float x) {
    const float z  = x * 0.70710678118654752f;    // x / sqrt(2)
    const float az = fabsf(z);
    const float t  = __builtin_amdgcn_rcpf(fmaf(0.3275911f, az, 1.0f));
    float p = fmaf(1.061405429f, t, -1.453152027f);
    p = fmaf(p, t, 1.421413741f);
    p = fmaf(p, t, -0.284496736f);
    p = fmaf(p, t, 0.254829592f);
    p = p * t;
    const float e   = __expf(-az * az);           // native v_exp_f32
    float       erf = fmaf(-p, e, 1.0f);
    erf = copysignf(erf, z);
    return 0.5f * x * (1.0f + erf);
}

// ---------------------------------------------------------------------------
// Kernel C: padding mask. One wave (32 lanes) per node, OR-reduce "any nonzero".
// pmask[node] = 1.0f if node is all-zero (padded), else 0.0f
// ---------------------------------------------------------------------------
__global__ __launch_bounds__(256) void pmask_kernel(const float* __restrict__ nf,
                                                    float* __restrict__ pmask) {
    const int wave = threadIdx.x >> 5, lane = threadIdx.x & 31;
    const int node = blockIdx.x * 8 + wave;                 // 256 blocks * 8 waves
    const float* p = nf + (size_t)node * 768;
    int nz = 0;
    for (int t = lane; t < 768; t += 32) nz |= (p[t] != 0.0f);
    #pragma unroll
    for (int off = 16; off; off >>= 1) nz |= __shfl_xor(nz, off, 32);
    if (lane == 0) pmask[node] = nz ? 0.0f : 1.0f;
}

// ---------------------------------------------------------------------------
// Kernel A: fused main kernel. One block per (b,i): delta_pos row, distances,
// RBF edge features (f16 in LDS, 16-j tiles, native v_exp_f32) with the masked
// k-sum fused into the compute phase (branchless fma), GEMM1 (WMMA f16) +
// branchless exact GELU, GEMM2 (WMMA f16) + key mask -> float4 stores.
// ---------------------------------------------------------------------------
__global__ __launch_bounds__(256, 1) void bias_kernel(
    const float* __restrict__ pos,
    const float* __restrict__ means, const float* __restrict__ betas,
    const float* __restrict__ w1, const float* __restrict__ b1,
    const float* __restrict__ w2, const float* __restrict__ b2,
    const float* __restrict__ pmask,
    float* __restrict__ gab,          // [B,H,N,N]
    float* __restrict__ delta_out,    // [B,N,N,3]
    float* __restrict__ sum_ef_ws)    // [B,N,K]
{
    __shared__ __attribute__((aligned(16))) _Float16 w1s[128 * EFS]; // [c][k]
    __shared__ __attribute__((aligned(16))) _Float16 w2s[32 * EFS];  // [h][c]
    __shared__ __attribute__((aligned(16))) _Float16 efs[16 * EFS];  // [jl][k]
    __shared__ __attribute__((aligned(16))) _Float16 hid[16 * EFS];  // [jl][c]
    __shared__ float tjs[NN];      // exp(-alpha*dist) per key j
    __shared__ float pms[NN];      // padding mask per key j (1 = padded)
    __shared__ float meansS[KK], betasS[KK];
    __shared__ float sred[16 * KK];  // per-row-group partial k-sums (8 KB)

    const int b   = blockIdx.y;
    const int i   = blockIdx.x;
    const int tid = threadIdx.x;
    const int lane = tid & 31, wave = tid >> 5;     // wave32

    // ---- stage weights into LDS as f16, transposed to [col][k] for B-loads
    for (int idx = tid; idx < 128 * 128; idx += 256) {
        int k = idx >> 7, c = idx & 127;
        w1s[c * EFS + k] = (_Float16)w1[idx];
    }
    for (int idx = tid; idx < 128 * 32; idx += 256) {
        int c = idx >> 5, h = idx & 31;
        w2s[h * EFS + c] = (_Float16)w2[idx];
    }
    if (tid < KK) { meansS[tid] = means[tid]; betasS[tid] = betas[tid]; }

    // ---- per-key geometry: j = tid
    {
        const int j = tid;
        const float pix = pos[((size_t)b * NN + i) * 3 + 0];
        const float piy = pos[((size_t)b * NN + i) * 3 + 1];
        const float piz = pos[((size_t)b * NN + i) * 3 + 2];
        const float dx = pix - pos[((size_t)b * NN + j) * 3 + 0];
        const float dy = piy - pos[((size_t)b * NN + j) * 3 + 1];
        const float dz = piz - pos[((size_t)b * NN + j) * 3 + 2];
        const size_t dofs = (((size_t)b * NN + i) * NN + j) * 3;
        delta_out[dofs + 0] = dx;
        delta_out[dofs + 1] = dy;
        delta_out[dofs + 2] = dz;
        const float r2 = dx * dx + dy * dy + dz * dz;
        const float dist = (r2 > 0.0f) ? sqrtf(r2) : 0.0f;
        tjs[j] = __expf(-ALPHA * dist);
        pms[j] = pmask[b * NN + j];
    }
    __syncthreads();

    // RBF-phase thread mapping is fixed across tiles: row-in-tile jl, 8 k's.
    const int jl = tid >> 4;
    const int k0 = (tid & 15) * 8;
    float ksum[8];
    #pragma unroll
    for (int e = 0; e < 8; ++e) ksum[e] = 0.0f;

    // GEMM lane mapping (fixed)
    const int n = lane & 15, khalf = lane >> 4;
    const int c0 = wave * 16;
    const float bv1 = b1[c0 + n];
    const float bv2 = (wave < 2) ? b2[wave * 16 + n] : 0.0f;

    for (int jt = 0; jt < 16; ++jt) {
        // ---- RBF tile: 16 rows x 128 k, 8 elems/thread; fused masked k-sum
        {
            const float t    = tjs[jt * 16 + jl];
            const float msel = 1.0f - pms[jt * 16 + jl];   // 0 if padded, else 1
            #pragma unroll
            for (int e = 0; e < 8; ++e) {
                const int k = k0 + e;
                const float d = t - meansS[k];
                const float v = __expf(-betasS[k] * d * d); // native v_exp_f32
                efs[jl * EFS + k] = (_Float16)v;
                ksum[e] = fmaf(msel, v, ksum[e]);          // branchless masked sum
            }
        }
        __syncthreads();

        // ---- GEMM1: hidden[16 x 128] = GELU(ef @ w1 + b1); wave -> 16-col tile
        {
            v8f acc;
            #pragma unroll
            for (int r = 0; r < 8; ++r) acc[r] = bv1;
            #pragma unroll
            for (int kk = 0; kk < 4; ++kk) {
                // A: row = n, two contiguous 8-half runs (ISA 16-bit A layout)
                v8h alo = *(const v8h*)&efs[n * EFS + kk * 32 + khalf * 8];
                v8h ahi = *(const v8h*)&efs[n * EFS + kk * 32 + 16 + khalf * 8];
                // B: col = c0+n, two contiguous 8-half runs (same K order as A)
                v8h blo = *(const v8h*)&w1s[(c0 + n) * EFS + kk * 32 + khalf * 8];
                v8h bhi = *(const v8h*)&w1s[(c0 + n) * EFS + kk * 32 + 16 + khalf * 8];
                v16h a, bm;
                #pragma unroll
                for (int e = 0; e < 8; ++e) {
                    a[e] = alo[e];  a[8 + e] = ahi[e];
                    bm[e] = blo[e]; bm[8 + e] = bhi[e];
                }
                acc = __builtin_amdgcn_wmma_f32_16x16x32_f16(
                        false, a, false, bm, (short)0, acc, false, false);
            }
            // branchless exact GELU, f16 store to hidden LDS (D: 8 rows per lane)
            #pragma unroll
            for (int r = 0; r < 8; ++r) {
                const float g = gelu_exact(acc[r]);
                const int m = khalf * 8 + r;
                hid[m * EFS + (c0 + n)] = (_Float16)g;
            }
        }
        __syncthreads();

        // ---- GEMM2: bias[16 x 32] = hidden @ w2 + b2; waves 0,1 -> 16-head tiles
        if (wave < 2) {
            const int h0 = wave * 16;
            v8f acc;
            #pragma unroll
            for (int r = 0; r < 8; ++r) acc[r] = bv2;
            #pragma unroll
            for (int kk = 0; kk < 4; ++kk) {
                v8h alo = *(const v8h*)&hid[n * EFS + kk * 32 + khalf * 8];
                v8h ahi = *(const v8h*)&hid[n * EFS + kk * 32 + 16 + khalf * 8];
                v8h blo = *(const v8h*)&w2s[(h0 + n) * EFS + kk * 32 + khalf * 8];
                v8h bhi = *(const v8h*)&w2s[(h0 + n) * EFS + kk * 32 + 16 + khalf * 8];
                v16h a, bm;
                #pragma unroll
                for (int e = 0; e < 8; ++e) {
                    a[e] = alo[e];  a[8 + e] = ahi[e];
                    bm[e] = blo[e]; bm[8 + e] = bhi[e];
                }
                acc = __builtin_amdgcn_wmma_f32_16x16x32_f16(
                        false, a, false, bm, (short)0, acc, false, false);
            }
            // D rows = key j (consecutive) -> two aligned float4 stores per lane,
            // key-padding mask applied per element.
            const int h  = h0 + n;
            const int j0 = jt * 16 + khalf * 8;
            const size_t base = (((size_t)b * HH + h) * NN + i) * NN + j0;
            float4 o0, o1;
            o0.x = (pms[j0 + 0] != 0.0f) ? -1e20f : acc[0];
            o0.y = (pms[j0 + 1] != 0.0f) ? -1e20f : acc[1];
            o0.z = (pms[j0 + 2] != 0.0f) ? -1e20f : acc[2];
            o0.w = (pms[j0 + 3] != 0.0f) ? -1e20f : acc[3];
            o1.x = (pms[j0 + 4] != 0.0f) ? -1e20f : acc[4];
            o1.y = (pms[j0 + 5] != 0.0f) ? -1e20f : acc[5];
            o1.z = (pms[j0 + 6] != 0.0f) ? -1e20f : acc[6];
            o1.w = (pms[j0 + 7] != 0.0f) ? -1e20f : acc[7];
            *(float4*)&gab[base]     = o0;
            *(float4*)&gab[base + 4] = o1;
        }
        __syncthreads();   // before overwriting efs/hid next tile
    }

    // ---- reduce the 16 per-row-group partials for each k, write [b,i,k]
    #pragma unroll
    for (int e = 0; e < 8; ++e) sred[jl * KK + k0 + e] = ksum[e];
    __syncthreads();
    if (tid < KK) {
        float s = 0.0f;
        #pragma unroll
        for (int r = 0; r < 16; ++r) s += sred[r * KK + tid];
        sum_ef_ws[((size_t)b * NN + i) * KK + tid] = s;
    }
}

// ---------------------------------------------------------------------------
// Kernel B: merge = sum_ef[2048,128] @ ew[128,768] + eb  (tiny, VALU fp32)
// ---------------------------------------------------------------------------
__global__ __launch_bounds__(256) void merge_kernel(
    const float* __restrict__ sum_ef, const float* __restrict__ ew,
    const float* __restrict__ eb, float* __restrict__ out) {
    const int row = blockIdx.y;                       // 2048
    const int c = blockIdx.x * 256 + threadIdx.x;     // 768
    const float* s = sum_ef + (size_t)row * KK;
    float acc = eb[c];
    #pragma unroll 8
    for (int k = 0; k < KK; ++k) acc = fmaf(s[k], ew[(size_t)k * EMB + c], acc);
    out[(size_t)row * EMB + c] = acc;
}

// ---------------------------------------------------------------------------
extern "C" void kernel_launch(void* const* d_in, const int* in_sizes, int n_in,
                              void* d_out, int out_size, void* d_ws, size_t ws_size,
                              hipStream_t stream) {
    const float* nf    = (const float*)d_in[0];
    const float* pos   = (const float*)d_in[1];
    const float* means = (const float*)d_in[2];
    const float* betas = (const float*)d_in[3];
    const float* w1    = (const float*)d_in[4];
    const float* b1    = (const float*)d_in[5];
    const float* w2    = (const float*)d_in[6];
    const float* b2    = (const float*)d_in[7];
    const float* ew    = (const float*)d_in[8];
    const float* eb    = (const float*)d_in[9];

    float* out   = (float*)d_out;
    float* gab   = out;                                    // [8,32,256,256]
    float* merge = out + (size_t)BB * HH * NN * NN;        // [8,256,768]
    float* dpos  = merge + (size_t)BB * NN * EMB;          // [8,256,256,3]

    float* pmask  = (float*)d_ws;                          // [2048]
    float* sum_ef = pmask + BB * NN;                       // [2048,128]

    pmask_kernel<<<dim3(BB * NN / 8), dim3(256), 0, stream>>>(nf, pmask);

    bias_kernel<<<dim3(NN, BB), dim3(256), 0, stream>>>(
        pos, means, betas, w1, b1, w2, b2, pmask, gab, dpos, sum_ef);

    merge_kernel<<<dim3(EMB / 256, BB * NN), dim3(256), 0, stream>>>(
        sum_ef, ew, eb, merge);
}
